// SelfAttention_2568390443459
// MI455X (gfx1250) — matrix operand order repaired and checked by reading, not compiled, not run on hardware
//
#include <hip/hip_runtime.h>

typedef __attribute__((ext_vector_type(16))) __bf16 v16bf;
typedef __attribute__((ext_vector_type(8)))  __bf16 v8bf;
typedef __attribute__((ext_vector_type(8)))  float  v8f;
typedef __attribute__((ext_vector_type(4)))  float  f32x4;

#define Bn   16
#define Dn   512
#define NHn  8
#define DHn  64
#define Ln   1024

// ---------------------------------------------------------------------------
// Kernel 1: fused projection GEMM  O[b] (1536 x 1024) = W (1536x512) @ X[b]
// X[b] is queries[b] in channels-first [D][L] layout (exactly the GEMM B op).
// Rows   0..511  -> K  stored [b][h][l][dh]      (bf16)
// Rows 512..1023 -> V  stored [b][h][dh][l]      (bf16, i.e. V^T)
// Rows 1024..1535-> Q  stored [b][h][l][dh]*1/8  (bf16, pre-scaled)
// Block: 256 thr = 8 waves, tile 32e x 64l, K-loop chunks of 32,
// async double-buffered X staging (global_load_async_to_lds_b128).
// ---------------------------------------------------------------------------
__global__ __launch_bounds__(256) void proj_kernel(
    const float* __restrict__ queries,
    const float* __restrict__ w_mem,
    const float* __restrict__ w_q,
    unsigned short* __restrict__ Ksu,
    unsigned short* __restrict__ Vsu,
    unsigned short* __restrict__ Qsu)
{
    __shared__ __align__(32) float  Xraw[2][32 * 64];   // raw f32 chunks (async dest)
    __shared__ __align__(32) __bf16 Xt[2][64 * 32];     // transposed bf16 [l][d]

    __bf16* Ks = reinterpret_cast<__bf16*>(Ksu);
    __bf16* Vs = reinterpret_cast<__bf16*>(Vsu);
    __bf16* Qs = reinterpret_cast<__bf16*>(Qsu);

    const int bx   = blockIdx.x;
    const int eblk = bx % 48;              // 48 tiles of 32 rows -> 1536
    const int lblk = (bx / 48) % 16;       // 16 tiles of 64 cols -> 1024
    const int b    = bx / (48 * 16);

    const int tid  = threadIdx.x;
    const int wid  = tid >> 5;
    const int lane = tid & 31;
    const int n16  = lane & 15;
    const int kh   = lane >> 4;

    const int ebase     = eblk * 32 + (wid >> 2) * 16;
    const int lbase_blk = lblk * 64;
    const int lw        = (wid & 3) * 16;

    const float* Xb = queries + (size_t)b * Dn * Ln;

    // A-fragment source row of W (concat of w_mem rows then w_q rows)
    const int erow = ebase + n16;
    const float* wrow = (erow < 1024) ? (w_mem + (size_t)erow * Dn)
                                      : (w_q   + (size_t)(erow - 1024) * Dn);

    // cooperative copy indexing: thread owns 8 floats (32B) of the 32x64 chunk
    const int dld  = tid >> 3;        // 0..31
    const int lseg = (tid & 7) * 8;   // 0..56

    auto issue_copy = [&](int buf, int kc) {
        unsigned long long g = (unsigned long long)(size_t)
            (Xb + (size_t)(kc + dld) * Ln + lbase_blk + lseg);
        unsigned int l = (unsigned int)(size_t)&Xraw[buf][dld * 64 + lseg];
        // INST_OFFSET applies to both global and LDS address (ISA 15.18.3)
        asm volatile("global_load_async_to_lds_b128 %0, %1, off"
                     :: "v"(l), "v"(g) : "memory");
        asm volatile("global_load_async_to_lds_b128 %0, %1, off offset:16"
                     :: "v"(l), "v"(g) : "memory");
    };

    issue_copy(0, 0);
    v8f acc = {};

    for (int kc = 0; kc < Dn; kc += 32) {
        const int cur = (kc >> 5) & 1;
        if (kc + 32 < Dn) {
            issue_copy(cur ^ 1, kc + 32);
            asm volatile("s_wait_asynccnt 0x2" ::: "memory");  // current chunk landed
        } else {
            asm volatile("s_wait_asynccnt 0x0" ::: "memory");
        }
        __syncthreads();                       // chunk visible to all waves

        // convert + transpose: Xraw[cur] (f32 [d][l]) -> Xt[cur] (bf16 [l][d])
        {
            f32x4 a0 = *(const f32x4*)&Xraw[cur][dld * 64 + lseg];
            f32x4 a1 = *(const f32x4*)&Xraw[cur][dld * 64 + lseg + 4];
#pragma unroll
            for (int i = 0; i < 4; ++i) Xt[cur][(lseg + i) * 32 + dld]     = (__bf16)a0[i];
#pragma unroll
            for (int i = 0; i < 4; ++i) Xt[cur][(lseg + 4 + i) * 32 + dld] = (__bf16)a1[i];
        }
        __syncthreads();                       // Xt[cur] ready

        // A fragment: 16 W-rows x 32 d (per-lane: row n16, d = 16*kh..+16 contiguous)
        union { v16bf v; v8bf h[2]; } A;
        {
            const float* p0 = wrow + kc + 16 * kh;
            const float* p1 = wrow + kc + 16 * kh + 8;
            f32x4 w00 = *(const f32x4*)p0, w01 = *(const f32x4*)(p0 + 4);
            f32x4 w10 = *(const f32x4*)p1, w11 = *(const f32x4*)(p1 + 4);
            v8bf h0, h1;
#pragma unroll
            for (int i = 0; i < 4; ++i) {
                h0[i]     = (__bf16)w00[i];  h0[4 + i] = (__bf16)w01[i];
                h1[i]     = (__bf16)w10[i];  h1[4 + i] = (__bf16)w11[i];
            }
            A.h[0] = h0; A.h[1] = h1;
        }

        // B fragment from LDS: column l = lw+n16, K-dim = 16*kh..+16 contiguous
        v16bf Bv = *(const v16bf*)&Xt[cur][(lw + n16) * 32 + 16 * kh];

        acc = __builtin_amdgcn_wmma_f32_16x16x32_bf16(false, A.v, false, Bv,
                                                      (short)0, acc, false, false);
    }

    // Epilogue: C layout -> lane n16 = l-col, VGPR r -> e-row r + 8*kh
    const int lglob = lbase_blk + lw + n16;
#pragma unroll
    for (int r = 0; r < 8; ++r) {
        const int e = ebase + r + 8 * kh;
        const float v = acc[r];
        if (e >= 1024) {                       // Q channels, pre-scale by DH^-0.5
            const int eq = e - 1024;
            Qs[((size_t)((b * NHn + (eq >> 6)) * Ln + lglob)) * DHn + (eq & 63)]
                = (__bf16)(v * 0.125f);
        } else if (e >= 512) {                 // V channels -> transposed [dh][l]
            const int ev = e - 512;
            Vs[((size_t)((b * NHn + (ev >> 6)) * DHn + (ev & 63))) * Ln + lglob]
                = (__bf16)v;
        } else {                               // K channels
            Ks[((size_t)((b * NHn + (e >> 6)) * Ln + lglob)) * DHn + (e & 63)]
                = (__bf16)v;
        }
    }
}

// ---------------------------------------------------------------------------
// Kernel 2: flash attention. Block = 256 thr = 8 waves; wave owns 16 q-rows.
// All 8 waves share (b,h): K/V 32-key tiles are staged ONCE per block into
// LDS with async-to-LDS copies (double-buffered, ASYNCcnt-tracked), killing
// the 8x redundant global reads and overlapping fetch i+1 with WMMAs of i.
// Per tile:  S^T = K_tile (A) @ Q^T (B)  -> in-lane softmax (+shfl_xor 16)
//            O^T += V^T (A) @ P^T (B)
// grid = B * NH * (L/128) = 1024 blocks.
// ---------------------------------------------------------------------------
__global__ __launch_bounds__(256) void attn_kernel(
    const unsigned short* __restrict__ Ksu,
    const unsigned short* __restrict__ Vsu,
    const unsigned short* __restrict__ Qsu,
    const int* __restrict__ mask,
    float* __restrict__ out)
{
    __shared__ float mbias[Ln];                          // 0 or -1e30 per key pos
    __shared__ __align__(32) __bf16 pbuf[8][16][32];     // per-wave P^T staging
    __shared__ __align__(32) __bf16 Kbuf[2][32 * 64];    // K tile [kpos][dh]
    __shared__ __align__(32) __bf16 Vbuf[2][64 * 32];    // V^T tile [dh][kpos]

    const __bf16* Ks = reinterpret_cast<const __bf16*>(Ksu);
    const __bf16* Vs = reinterpret_cast<const __bf16*>(Vsu);
    const __bf16* Qs = reinterpret_cast<const __bf16*>(Qsu);

    const int bx   = blockIdx.x;
    const int qblk = bx & 7;
    const int h    = (bx >> 3) & 7;
    const int b    = bx >> 6;

    const int tid  = threadIdx.x;
    const int wid  = tid >> 5;
    const int lane = tid & 31;
    const int n16  = lane & 15;
    const int kh   = lane >> 4;

    const size_t bh = (size_t)(b * NHn + h);
    const __bf16* Kb = Ks + bh * Ln * DHn;     // [l][dh], tile rows contiguous
    const __bf16* Vb = Vs + bh * DHn * Ln;     // [dh][l]
    const __bf16* Qb = Qs + bh * Ln * DHn;

    // async tile copy: K tile is one contiguous 4KB block; V tile is 64x64B rows.
    auto issue_copy = [&](int buf, int kb) {
        unsigned long long gk = (unsigned long long)(size_t)(Kb + (size_t)kb * DHn + tid * 8);
        unsigned int lk = (unsigned int)(size_t)&Kbuf[buf][tid * 8];
        asm volatile("global_load_async_to_lds_b128 %0, %1, off"
                     :: "v"(lk), "v"(gk) : "memory");
        unsigned long long gv = (unsigned long long)(size_t)
            (Vb + (size_t)(tid >> 2) * Ln + kb + (tid & 3) * 8);
        unsigned int lv = (unsigned int)(size_t)&Vbuf[buf][(tid >> 2) * 32 + (tid & 3) * 8];
        asm volatile("global_load_async_to_lds_b128 %0, %1, off"
                     :: "v"(lv), "v"(gv) : "memory");
    };

    issue_copy(0, 0);                          // prefetch first tile

    for (int i = tid; i < Ln; i += 256)
        mbias[i] = mask[b * Ln + i] ? 0.0f : -1.0e30f;

    const int qbase = qblk * 128 + wid * 16;

    // Loop-invariant Q B-fragments (column q = n16, K-dim d contiguous)
    v16bf fq0, fq1;
    {
        const __bf16* qrow = Qb + (size_t)(qbase + n16) * DHn;
        fq0 = *(const v16bf*)(qrow + 16 * kh);         // d-chunk 0..31
        fq1 = *(const v16bf*)(qrow + 32 + 16 * kh);    // d-chunk 32..63
    }

    v8f accO[4] = { {}, {}, {}, {} };   // O^T tiles, dh-blocks 0..3
    float mrun = -3.0e38f, srun = 0.0f;

    for (int kb = 0; kb < Ln; kb += 32) {
        const int cur = (kb >> 5) & 1;
        if (kb + 32 < Ln) {
            issue_copy(cur ^ 1, kb + 32);                      // prefetch next tile
            asm volatile("s_wait_asynccnt 0x2" ::: "memory");  // current tile landed
        } else {
            asm volatile("s_wait_asynccnt 0x0" ::: "memory");
        }
        __syncthreads();               // tile (and, first iter, mbias) visible

        // ---- S^T = K @ Q^T : two 16-kpos tiles, K-dim 64 in two chunks ----
        v8f c[2];
#pragma unroll
        for (int t = 0; t < 2; ++t) {
            const __bf16* krow = &Kbuf[cur][(16 * t + n16) * DHn];
            union { v16bf v; v8bf h[2]; } a0, a1;
            a0.h[0] = *(const v8bf*)(krow + 16 * kh);
            a0.h[1] = *(const v8bf*)(krow + 8 + 16 * kh);
            a1.h[0] = *(const v8bf*)(krow + 32 + 16 * kh);
            a1.h[1] = *(const v8bf*)(krow + 40 + 16 * kh);
            v8f cz = {};
            cz = __builtin_amdgcn_wmma_f32_16x16x32_bf16(false, a0.v, false, fq0,
                                                         (short)0, cz, false, false);
            cz = __builtin_amdgcn_wmma_f32_16x16x32_bf16(false, a1.v, false, fq1,
                                                         (short)0, cz, false, false);
            c[t] = cz;
        }

        // ---- online softmax (reduction over kpos lives in-lane + xor16) ----
        float sv[16];
#pragma unroll
        for (int r = 0; r < 8; ++r) {
            sv[r]     = c[0][r] + mbias[kb + 8 * kh + r];
            sv[8 + r] = c[1][r] + mbias[kb + 16 + 8 * kh + r];
        }
        float ml = sv[0];
#pragma unroll
        for (int i = 1; i < 16; ++i) ml = fmaxf(ml, sv[i]);
        ml = fmaxf(ml, __shfl_xor(ml, 16, 32));
        const float mnew = fmaxf(mrun, ml);
        const float corr = __expf(mrun - mnew);
        float ps = 0.0f;
#pragma unroll
        for (int i = 0; i < 16; ++i) {
            const float p = __expf(sv[i] - mnew);
            sv[i] = p; ps += p;
        }
        ps += __shfl_xor(ps, 16, 32);
        srun = srun * corr + ps;
        mrun = mnew;
#pragma unroll
        for (int d = 0; d < 4; ++d)
#pragma unroll
            for (int r = 0; r < 8; ++r) accO[d][r] *= corr;

        // ---- restage P^T through wave-private LDS: [q][kpos 0..31] bf16 ----
        v8bf p0, p1;
#pragma unroll
        for (int r = 0; r < 8; ++r) { p0[r] = (__bf16)sv[r]; p1[r] = (__bf16)sv[8 + r]; }
        *(v8bf*)&pbuf[wid][n16][8 * kh]      = p0;   // kpos 8*kh..+8    (tile 0)
        *(v8bf*)&pbuf[wid][n16][16 + 8 * kh] = p1;   // kpos 16+8*kh..+8 (tile 1)
        __builtin_amdgcn_wave_barrier();             // DS is in-order per wave
        v16bf fp = *(const v16bf*)&pbuf[wid][n16][16 * kh]; // B-frag: col q, 16 kpos
        __builtin_amdgcn_wave_barrier();

        // ---- O^T += V^T @ P^T, four dh-blocks of 16 ----
#pragma unroll
        for (int dhb = 0; dhb < 4; ++dhb) {
            const __bf16* vrow = &Vbuf[cur][(dhb * 16 + n16) * 32];
            union { v16bf v; v8bf h[2]; } av;
            av.h[0] = *(const v8bf*)(vrow + 16 * kh);
            av.h[1] = *(const v8bf*)(vrow + 8 + 16 * kh);
            accO[dhb] = __builtin_amdgcn_wmma_f32_16x16x32_bf16(false, av.v, false, fp,
                                                                (short)0, accO[dhb],
                                                                false, false);
        }
        __syncthreads();               // all waves done with tile before overwrite
    }

    // ---- normalize and store: O^T layout == channels-first output layout ----
    const float inv = 1.0f / srun;
    const size_t obase = ((size_t)b * Dn + h * DHn) * Ln + qbase + n16;
#pragma unroll
    for (int dhb = 0; dhb < 4; ++dhb)
#pragma unroll
        for (int r = 0; r < 8; ++r) {
            const int dh = dhb * 16 + r + 8 * kh;
            out[obase + (size_t)dh * Ln] = accO[dhb][r] * inv;
        }
}

// ---------------------------------------------------------------------------
extern "C" void kernel_launch(void* const* d_in, const int* in_sizes, int n_in,
                              void* d_out, int out_size, void* d_ws, size_t ws_size,
                              hipStream_t stream) {
    const float* queries = (const float*)d_in[0];
    const int*   mask    = (const int*)d_in[1];
    const float* w_mem   = (const float*)d_in[2];
    const float* w_q     = (const float*)d_in[3];
    float* out = (float*)d_out;

    const size_t per = (size_t)Bn * NHn * Ln * DHn;       // 8M bf16 elems each
    unsigned short* Ks = (unsigned short*)d_ws;
    unsigned short* Vs = Ks + per;
    unsigned short* Qs = Vs + per;

    proj_kernel<<<dim3(48 * 16 * Bn), 256, 0, stream>>>(queries, w_mem, w_q, Ks, Vs, Qs);
    attn_kernel<<<dim3(Bn * NHn * (Ln / 128)), 256, 0, stream>>>(Ks, Vs, Qs, mask, out);
}
